// LSTMDecoderG_77300821393610
// MI455X (gfx1250) — matrix operand order repaired and checked by reading, not compiled, not run on hardware
//
#include <hip/hip_runtime.h>
#include <hip/hip_bf16.h>
#include <math.h>

// ---------------- problem constants ----------------
constexpr int V  = 20000, E = 512, H = 1024, NL = 2, NH = 4, LE = 64;
constexpr int B  = 64, T = 12, K = 36, L = 20, MT = 4;
constexpr int DH = H / NH;                 // 256
constexpr int LSTM_IN = E + 2 * H + LE;    // 2624
constexpr int PGN_IN  = 2 * H + LSTM_IN;   // 4672
constexpr float SCALE = 0.0625f;           // 1/sqrt(DH)

typedef __bf16 bf16_t;
typedef __attribute__((ext_vector_type(16))) __bf16 bf16x16;
typedef __attribute__((ext_vector_type(8)))  __bf16 bf16x8;
typedef __attribute__((ext_vector_type(8)))  float  floatx8;

__device__ __forceinline__ float sigmoidf_(float x) { return 1.f / (1.f + __expf(-x)); }

// ---------------- fp32 -> bf16 conversion (weights once, activations per GEMM) ----------------
__global__ __launch_bounds__(256)
void cvt_f32_bf16(const float* __restrict__ src, bf16_t* __restrict__ dst, size_t n) {
    size_t i = (size_t)blockIdx.x * 256 + threadIdx.x;
    if (i < n) dst[i] = (bf16_t)src[i];
}

// ---------------- WMMA GEMM: C(MxN) = A(MxK) @ W(NxK)^T (+bias)(+=C) ----------------
// A bf16 row-major (pre-converted), W bf16 row-major (PyTorch layout), C fp32 row-major.
// Each wave owns one 16-wide N tile and FOUR 16-row M tiles (M%64==0): a single B
// fragment load feeds 4 independent v_wmma_f32_16x16x32_bf16 -> weight bytes are
// touched exactly once per GEMM. Fragment K order (wave32): contiguous, k = 16*half + s, same in A and B.
__global__ __launch_bounds__(128)
void gemm_bf16(const bf16_t* __restrict__ A, const bf16_t* __restrict__ W,
               const float* __restrict__ bias, float* __restrict__ C,
               int M, int N, int Kd, int accumulate) {
    const int lane = threadIdx.x & 31;
    const int wave = threadIdx.x >> 5;
    const int n0 = (blockIdx.x * 4 + wave) * 16;
    const int m0 = blockIdx.y * 64;
    if (n0 >= N || m0 >= M) return;            // wave-uniform: EXEC all-ones for WMMA

    const int nh   = lane & 15;                // B/C/D: N = lane%16 ; A: M = lane%16
    const int half = lane >> 4;                // 0: lanes 0-15, 1: lanes 16-31
    const int n    = n0 + nh;
    const int kaof = half * 16;                // A upper lanes: K offset +16
    const int kbof = half * 16;                // B upper lanes: K offset +16

    floatx8 acc[4];
    {
        const float binit = bias ? bias[n] : 0.0f;
#pragma unroll
        for (int mt = 0; mt < 4; ++mt) {
            const int mb = m0 + mt * 16 + half * 8;   // C/D: upper lanes hold M+8
#pragma unroll
            for (int r = 0; r < 8; ++r)
                acc[mt][r] = (accumulate ? C[(size_t)(mb + r) * N + n] : 0.0f) + binit;
        }
    }

    const bf16_t* Wrow = W + (size_t)n * Kd;
    const bf16_t* A0   = A + (size_t)(m0 + nh) * Kd + kaof;
    const size_t  Astr = (size_t)16 * Kd;      // stride between M tiles

    for (int k0 = 0; k0 < Kd; k0 += 32) {
        // B fragment 32x16: 16 contiguous K values of weight row n (32B load)
        bf16x16 b = *reinterpret_cast<const bf16x16*>(Wrow + k0 + kbof);
        if (k0 + 32 < Kd) __builtin_prefetch(Wrow + k0 + 32 + kbof, 0, 1);
#pragma unroll
        for (int mt = 0; mt < 4; ++mt) {
            // A fragment 16x32: elems 0..15 <- K=k0+kaof.. contiguous
            const bf16_t* ap = A0 + (size_t)mt * Astr + k0;
            bf16x8 lo = *reinterpret_cast<const bf16x8*>(ap);
            bf16x8 hi = *reinterpret_cast<const bf16x8*>(ap + 8);
            bf16x16 a;
#pragma unroll
            for (int e = 0; e < 8; ++e) { a[e] = lo[e]; a[8 + e] = hi[e]; }
            acc[mt] = __builtin_amdgcn_wmma_f32_16x16x32_bf16(
                          false, a, false, b, (short)0, acc[mt], false, false);
        }
    }
#pragma unroll
    for (int mt = 0; mt < 4; ++mt) {
        const int mb = m0 + mt * 16 + half * 8;
#pragma unroll
        for (int r = 0; r < 8; ++r)
            C[(size_t)(mb + r) * N + n] = acc[mt][r];
    }
}

// ---------------- fused per-head attention (scores+softmax+context) ----------------
__global__ __launch_bounds__(256)
void attn_kernel(const float* __restrict__ q,     // (B,H) head-concat
                 const float* __restrict__ kk,    // (B,S,H)
                 const float* __restrict__ vv,    // (B,S,H)
                 const unsigned char* __restrict__ mask, // (B,S) or null
                 float* __restrict__ ctx,         // (B,H)
                 float* __restrict__ alpha,       // (B,NH,S)
                 int S) {
    const int b = blockIdx.x, h = blockIdx.y, t = threadIdx.x;
    __shared__ float qs[DH];
    __shared__ float sc[64];
    qs[t] = q[(size_t)b * H + h * DH + t];
    __syncthreads();
    if (t < S) {
        const float* kp = kk + (size_t)(b * S + t) * H + h * DH;
        float s = 0.f;
#pragma unroll 8
        for (int d = 0; d < DH; ++d) s += qs[d] * kp[d];
        s *= SCALE;
        if (mask && !mask[b * S + t]) s = -1e9f;
        sc[t] = s;
    }
    __syncthreads();
    if (t == 0) {
        float mx = -3.4e38f;
        for (int s = 0; s < S; ++s) mx = fmaxf(mx, sc[s]);
        float sum = 0.f;
        for (int s = 0; s < S; ++s) { float e = __expf(sc[s] - mx); sc[s] = e; sum += e; }
        const float inv = 1.f / sum;
        for (int s = 0; s < S; ++s) {
            sc[s] *= inv;
            alpha[((size_t)b * NH + h) * S + s] = sc[s];
        }
    }
    __syncthreads();
    float a = 0.f;
    for (int s = 0; s < S; ++s)
        a += sc[s] * vv[(size_t)(b * S + s) * H + h * DH + t];
    ctx[(size_t)b * H + h * DH + t] = a;
}

// ---------------- build x = [embed_t | img_ctx | q_ctx | len_emb] ----------------
__global__ __launch_bounds__(256)
void build_x(const float* __restrict__ embW, const int* __restrict__ tseq, int t,
             const float* __restrict__ img_ctx, const float* __restrict__ q_ctx,
             const float* __restrict__ lenW, const int* __restrict__ lbin,
             float* __restrict__ x) {
    const int b = blockIdx.x;
    for (int j = threadIdx.x; j < LSTM_IN; j += blockDim.x) {
        float v;
        if (j < E)              v = embW[(size_t)tseq[b * T + t] * E + j];
        else if (j < E + H)     v = img_ctx[(size_t)b * H + (j - E)];
        else if (j < E + 2 * H) v = q_ctx[(size_t)b * H + (j - E - H)];
        else                    v = lenW[(size_t)lbin[b] * LE + (j - E - 2 * H)];
        x[(size_t)b * LSTM_IN + j] = v;
    }
}

// ---------------- LSTM pointwise (gate order i,f,g,o) ----------------
__global__ __launch_bounds__(256)
void lstm_ew(const float* __restrict__ g, float* __restrict__ c, float* __restrict__ h) {
    const int b = blockIdx.x;
    const float* gr = g + (size_t)b * 4 * H;
    for (int j = threadIdx.x; j < H; j += blockDim.x) {
        const float i  = sigmoidf_(gr[j]);
        const float f  = sigmoidf_(gr[H + j]);
        const float gg = tanhf(gr[2 * H + j]);
        const float o  = sigmoidf_(gr[3 * H + j]);
        const float cn = f * c[(size_t)b * H + j] + i * gg;
        c[(size_t)b * H + j] = cn;
        h[(size_t)b * H + j] = o * tanhf(cn);
    }
}

// ---------------- pointer-gate: raw logits (B,3) over 4672-dim concat ----------------
__global__ __launch_bounds__(256)
void pgn_dot(const float* __restrict__ img_ctx, const float* __restrict__ h_top,
             const float* __restrict__ x, const float* __restrict__ pgnW,
             const float* __restrict__ pgnb, float* __restrict__ raw) {
    const int b = blockIdx.x, j = blockIdx.y, t = threadIdx.x;
    __shared__ float red[256];
    float acc = 0.f;
    for (int i = t; i < PGN_IN; i += 256) {
        float xin;
        if (i < H)          xin = img_ctx[(size_t)b * H + i];
        else if (i < 2 * H) xin = h_top[(size_t)b * H + (i - H)];
        else                xin = x[(size_t)b * LSTM_IN + (i - 2 * H)];
        acc += xin * pgnW[(size_t)j * PGN_IN + i];
    }
    red[t] = acc; __syncthreads();
    for (int s = 128; s > 0; s >>= 1) { if (t < s) red[t] += red[t + s]; __syncthreads(); }
    if (t == 0) raw[b * 3 + j] = red[0] + pgnb[j];
}

__global__ void pgn_softmax(const float* __restrict__ raw, float* __restrict__ pw) {
    const int b = threadIdx.x; // blockDim == B
    const float a0 = raw[b * 3], a1 = raw[b * 3 + 1], a2 = raw[b * 3 + 2];
    const float mx = fmaxf(a0, fmaxf(a1, a2));
    const float e0 = __expf(a0 - mx), e1 = __expf(a1 - mx), e2 = __expf(a2 - mx);
    const float inv = 1.f / (e0 + e1 + e2);
    pw[b * 3] = e0 * inv; pw[b * 3 + 1] = e1 * inv; pw[b * 3 + 2] = e2 * inv;
}

// ---------------- row softmax over V ----------------
__global__ __launch_bounds__(256)
void softmax_row(const float* __restrict__ in, float* __restrict__ out, int n) {
    const int b = blockIdx.x, t = threadIdx.x;
    __shared__ float red[256];
    const float* row = in + (size_t)b * n;
    float*      orow = out + (size_t)b * n;
    float mx = -3.4e38f;
    for (int i = t; i < n; i += 256) mx = fmaxf(mx, row[i]);
    red[t] = mx; __syncthreads();
    for (int s = 128; s > 0; s >>= 1) { if (t < s) red[t] = fmaxf(red[t], red[t + s]); __syncthreads(); }
    mx = red[0]; __syncthreads();
    float sum = 0.f;
    for (int i = t; i < n; i += 256) { float e = __expf(row[i] - mx); orow[i] = e; sum += e; }
    red[t] = sum; __syncthreads();
    for (int s = 128; s > 0; s >>= 1) { if (t < s) red[t] += red[t + s]; __syncthreads(); }
    const float inv = 1.f / red[0];
    for (int i = t; i < n; i += 256) orow[i] *= inv;
}

// ---------------- p = pw0*pv, then scatter copy dists, then log ----------------
__global__ __launch_bounds__(256)
void mix_init(const float* __restrict__ pv, const float* __restrict__ pw,
              float* __restrict__ acc) {
    size_t i = (size_t)blockIdx.x * 256 + threadIdx.x;
    if (i < (size_t)B * V) {
        const int b = (int)(i / V);
        acc[i] = pw[b * 3] * pv[i];
    }
}

__global__ __launch_bounds__(256)
void scatter_copy(const int* __restrict__ qtok, const float* __restrict__ alphaQ,  // (B,NH,L)
                  const int* __restrict__ labels, const float* __restrict__ alphaI,// (B,NH,K)
                  const float* __restrict__ pw, float* __restrict__ acc) {
    const int idx = blockIdx.x * 256 + threadIdx.x;
    const int nq = B * L;
    if (idx < nq) {
        const int b = idx / L, l = idx % L;
        float a = 0.f;
        for (int h = 0; h < NH; ++h) a += alphaQ[((size_t)b * NH + h) * L + l];
        a *= (1.0f / NH);
        atomicAdd(&acc[(size_t)b * V + qtok[b * L + l]], pw[b * 3 + 1] * a);
    } else if (idx < nq + B * K * MT) {
        const int r = idx - nq;
        const int b = r / (K * MT), km = r % (K * MT), k = km / MT;
        float a = 0.f;
        for (int h = 0; h < NH; ++h) a += alphaI[((size_t)b * NH + h) * K + k];
        a *= (1.0f / (NH * MT));
        atomicAdd(&acc[(size_t)b * V + labels[(size_t)b * K * MT + km]], pw[b * 3 + 2] * a);
    }
}

__global__ __launch_bounds__(256)
void log_write(const float* __restrict__ acc, float* __restrict__ out, int t) {
    size_t i = (size_t)blockIdx.x * 256 + threadIdx.x;
    if (i < (size_t)B * V) {
        const int b = (int)(i / V), v = (int)(i % V);
        out[((size_t)b * T + t) * V + v] = logf(acc[i] + 1e-9f);
    }
}

__global__ void init_state(const float* __restrict__ eh, const float* __restrict__ ec,
                           float* __restrict__ hs, float* __restrict__ cs, size_t n) {
    size_t i = (size_t)blockIdx.x * 256 + threadIdx.x;
    if (i < n) { hs[i] = eh[i]; cs[i] = ec[i]; }
}

__global__ void write_scalar0(float* __restrict__ out) { out[(size_t)B * T * V] = 0.0f; }

// ---------------- host orchestration ----------------
static inline size_t alignup(size_t x) { return (x + 255) & ~(size_t)255; }

// convert A fp32 -> bf16 scratch (ordered on stream), then WMMA GEMM
static void launch_gemm(const float* A, bf16_t* aScratch,
                        const bf16_t* W, const float* bias, float* C,
                        int M, int N, int Kd, int accum, hipStream_t s) {
    const size_t aElems = (size_t)M * Kd;
    cvt_f32_bf16<<<dim3((unsigned)((aElems + 255) / 256)), dim3(256), 0, s>>>(A, aScratch, aElems);
    dim3 grid((N + 63) / 64, M / 64);
    gemm_bf16<<<grid, dim3(128), 0, s>>>(aScratch, W, bias, C, M, N, Kd, accum);
}

extern "C" void kernel_launch(void* const* d_in, const int* in_sizes, int n_in,
                              void* d_out, int out_size, void* d_ws, size_t ws_size,
                              hipStream_t stream) {
    (void)in_sizes; (void)n_in; (void)out_size; (void)ws_size;
    const float* encoder_h  = (const float*)d_in[0];
    const float* encoder_c  = (const float*)d_in[1];
    const float* img_feat   = (const float*)d_in[2];
    const float* q_hidden   = (const float*)d_in[3];
    const int*   target_seq = (const int*)d_in[4];
    const int*   q_tok      = (const int*)d_in[5];
    const unsigned char* img_mask = (const unsigned char*)d_in[6];
    const int*   length_bin = (const int*)d_in[7];
    const int*   labels     = (const int*)d_in[8];
    const float* emb_W      = (const float*)d_in[9];
    const float* len_W      = (const float*)d_in[10];
    const float* img_Wq = (const float*)d_in[11]; const float* img_bq = (const float*)d_in[12];
    const float* img_Wk = (const float*)d_in[13]; const float* img_bk = (const float*)d_in[14];
    const float* img_Wv = (const float*)d_in[15]; const float* img_bv = (const float*)d_in[16];
    const float* img_Wo = (const float*)d_in[17]; const float* img_bo = (const float*)d_in[18];
    const float* qa_Wq  = (const float*)d_in[19]; const float* qa_bq  = (const float*)d_in[20];
    const float* qa_Wk  = (const float*)d_in[21]; const float* qa_bk  = (const float*)d_in[22];
    const float* qa_Wv  = (const float*)d_in[23]; const float* qa_bv  = (const float*)d_in[24];
    const float* qa_Wo  = (const float*)d_in[25]; const float* qa_bo  = (const float*)d_in[26];
    const float* l0_Wih = (const float*)d_in[27]; const float* l0_Whh = (const float*)d_in[28];
    const float* l0_bih = (const float*)d_in[29]; const float* l0_bhh = (const float*)d_in[30];
    const float* l1_Wih = (const float*)d_in[31]; const float* l1_Whh = (const float*)d_in[32];
    const float* l1_bih = (const float*)d_in[33]; const float* l1_bhh = (const float*)d_in[34];
    const float* out_W  = (const float*)d_in[35]; const float* out_b  = (const float*)d_in[36];
    const float* pgn_W  = (const float*)d_in[37]; const float* pgn_b  = (const float*)d_in[38];
    float* out = (float*)d_out;

    // ---- workspace carve-up ----
    char* base = (char*)d_ws;
    size_t off = 0;
    auto bump = [&](size_t bytes) { void* p = base + off; off = alignup(off + bytes); return p; };
    auto bumpBF = [&](size_t n) { return (bf16_t*)bump(n * sizeof(bf16_t)); };
    auto bumpF  = [&](size_t n) { return (float*)bump(n * sizeof(float)); };

    bf16_t* wImgQ = bumpBF((size_t)H * H); bf16_t* wImgK = bumpBF((size_t)H * H);
    bf16_t* wImgV = bumpBF((size_t)H * H); bf16_t* wImgO = bumpBF((size_t)H * H);
    bf16_t* wQaQ  = bumpBF((size_t)H * H); bf16_t* wQaK  = bumpBF((size_t)H * H);
    bf16_t* wQaV  = bumpBF((size_t)H * H); bf16_t* wQaO  = bumpBF((size_t)H * H);
    bf16_t* wL0ih = bumpBF((size_t)4 * H * LSTM_IN);
    bf16_t* wL0hh = bumpBF((size_t)4 * H * H);
    bf16_t* wL1ih = bumpBF((size_t)4 * H * H);
    bf16_t* wL1hh = bumpBF((size_t)4 * H * H);
    bf16_t* wOut  = bumpBF((size_t)E * H);
    bf16_t* wEmb  = bumpBF((size_t)V * E);
    bf16_t* aScr  = bumpBF((size_t)B * K * H);   // activation bf16 scratch (max M*Kd)

    float* kImg = bumpF((size_t)B * K * H); float* vImg = bumpF((size_t)B * K * H);
    float* kQ   = bumpF((size_t)B * L * H); float* vQ   = bumpF((size_t)B * L * H);
    float* hs   = bumpF((size_t)NL * B * H);  // h0 | h1
    float* cs   = bumpF((size_t)NL * B * H);  // c0 | c1
    float* qbuf   = bumpF((size_t)B * H);
    float* ctxpre = bumpF((size_t)B * H);
    float* imgCtx = bumpF((size_t)B * H);
    float* qCtx   = bumpF((size_t)B * H);
    float* alphaI = bumpF((size_t)B * NH * K);
    float* alphaQ = bumpF((size_t)B * NH * L);
    float* xbuf   = bumpF((size_t)B * LSTM_IN);
    float* gbuf   = bumpF((size_t)B * 4 * H);
    float* proj   = bumpF((size_t)B * E);
    float* vlogit = bumpF((size_t)B * V);
    float* pv     = bumpF((size_t)B * V);
    float* accb   = bumpF((size_t)B * V);
    float* pwraw  = bumpF((size_t)B * 3);
    float* pw     = bumpF((size_t)B * 3);

    // ---- one-time (per launch): weight conversion to bf16 ----
    auto cvt = [&](const float* s, bf16_t* d, size_t n) {
        cvt_f32_bf16<<<dim3((unsigned)((n + 255) / 256)), dim3(256), 0, stream>>>(s, d, n);
    };
    cvt(img_Wq, wImgQ, (size_t)H * H); cvt(img_Wk, wImgK, (size_t)H * H);
    cvt(img_Wv, wImgV, (size_t)H * H); cvt(img_Wo, wImgO, (size_t)H * H);
    cvt(qa_Wq,  wQaQ,  (size_t)H * H); cvt(qa_Wk,  wQaK,  (size_t)H * H);
    cvt(qa_Wv,  wQaV,  (size_t)H * H); cvt(qa_Wo,  wQaO,  (size_t)H * H);
    cvt(l0_Wih, wL0ih, (size_t)4 * H * LSTM_IN);
    cvt(l0_Whh, wL0hh, (size_t)4 * H * H);
    cvt(l1_Wih, wL1ih, (size_t)4 * H * H);
    cvt(l1_Whh, wL1hh, (size_t)4 * H * H);
    cvt(out_W,  wOut,  (size_t)E * H);
    cvt(emb_W,  wEmb,  (size_t)V * E);

    // ---- init state and hoisted K/V projections ----
    {
        const size_t n = (size_t)NL * B * H;
        init_state<<<dim3((unsigned)((n + 255) / 256)), dim3(256), 0, stream>>>(
            encoder_h, encoder_c, hs, cs, n);
    }
    launch_gemm(img_feat, aScr, wImgK, img_bk, kImg, B * K, H, H, 0, stream);
    launch_gemm(img_feat, aScr, wImgV, img_bv, vImg, B * K, H, H, 0, stream);
    launch_gemm(q_hidden, aScr, wQaK,  qa_bk, kQ,   B * L, H, H, 0, stream);
    launch_gemm(q_hidden, aScr, wQaV,  qa_bv, vQ,   B * L, H, H, 0, stream);

    float* h0 = hs;             float* h1 = hs + (size_t)B * H;
    float* c0 = cs;             float* c1 = cs + (size_t)B * H;
    const unsigned bvBlocks = (unsigned)(((size_t)B * V + 255) / 256);

    // ---- sequential decode loop ----
    for (int t = 0; t < T; ++t) {
        // image attention
        launch_gemm(h1, aScr, wImgQ, img_bq, qbuf, B, H, H, 0, stream);
        attn_kernel<<<dim3(B, NH), dim3(DH), 0, stream>>>(qbuf, kImg, vImg, img_mask,
                                                          ctxpre, alphaI, K);
        launch_gemm(ctxpre, aScr, wImgO, img_bo, imgCtx, B, H, H, 0, stream);
        // question attention
        launch_gemm(h1, aScr, wQaQ, qa_bq, qbuf, B, H, H, 0, stream);
        attn_kernel<<<dim3(B, NH), dim3(DH), 0, stream>>>(qbuf, kQ, vQ, nullptr,
                                                          ctxpre, alphaQ, L);
        launch_gemm(ctxpre, aScr, wQaO, qa_bo, qCtx, B, H, H, 0, stream);
        // x and pointer gate (needs pre-update h1)
        build_x<<<dim3(B), dim3(256), 0, stream>>>(emb_W, target_seq, t, imgCtx, qCtx,
                                                   len_W, length_bin, xbuf);
        pgn_dot<<<dim3(B, 3), dim3(256), 0, stream>>>(imgCtx, h1, xbuf, pgn_W, pgn_b, pwraw);
        pgn_softmax<<<dim3(1), dim3(B), 0, stream>>>(pwraw, pw);
        // LSTM layer 0
        launch_gemm(xbuf, aScr, wL0ih, l0_bih, gbuf, B, 4 * H, LSTM_IN, 0, stream);
        launch_gemm(h0,   aScr, wL0hh, l0_bhh, gbuf, B, 4 * H, H, 1, stream);
        lstm_ew<<<dim3(B), dim3(256), 0, stream>>>(gbuf, c0, h0);
        // LSTM layer 1
        launch_gemm(h0, aScr, wL1ih, l1_bih, gbuf, B, 4 * H, H, 0, stream);
        launch_gemm(h1, aScr, wL1hh, l1_bhh, gbuf, B, 4 * H, H, 1, stream);
        lstm_ew<<<dim3(B), dim3(256), 0, stream>>>(gbuf, c1, h1);
        // tied output head: (B,E) then (B,V)
        launch_gemm(h1,   aScr, wOut, out_b,  proj,   B, E, H, 0, stream);
        launch_gemm(proj, aScr, wEmb, nullptr, vlogit, B, V, E, 0, stream);
        softmax_row<<<dim3(B), dim3(256), 0, stream>>>(vlogit, pv, V);
        // mix + copy distributions + log
        mix_init<<<dim3(bvBlocks), dim3(256), 0, stream>>>(pv, pw, accb);
        scatter_copy<<<dim3((B * L + B * K * MT + 255) / 256), dim3(256), 0, stream>>>(
            q_tok, alphaQ, labels, alphaI, pw, accb);
        log_write<<<dim3(bvBlocks), dim3(256), 0, stream>>>(accb, out, t);
    }
    // trailing scalar output (second tuple element)
    write_scalar0<<<dim3(1), dim3(1), 0, stream>>>(out);
}
